// SimpleGCN_15461882266039
// MI455X (gfx1250) — compile-verified, matrix operand order checked by reading, not run on hardware
//
#include <hip/hip_runtime.h>
#include <hip/hip_bf16.h>
#include <stdint.h>

// ---------------- CDNA5 WMMA types ----------------
typedef __bf16 v16bf __attribute__((ext_vector_type(16)));
typedef float  v8f   __attribute__((ext_vector_type(8)));

#define NNODE 8192
#define INCH  512
#define HIDCH 128
#define OUTCH 64
#define MTILES 512   // NNODE/16 (WMMA M)
#define KTILES 256   // NNODE/32 (WMMA K for bf16)

// ---- tile element addressing (ISA 16-bit A 16x32 / mirrored B 32x16 layout) ----
// lane = m_or_n + 16*((k>>3)&1) ; half-in-lane = (k&7) + 8*(k>=16) ; 16 halfs/lane.
__device__ __forceinline__ size_t tile_off(int mn, int kin) {
    int lane = mn + (((kin >> 3) & 1) << 4);
    int hh   = (kin & 7) + (((kin >> 4) & 1) << 3);
    return (size_t)lane * 16 + hh;
}

// =====================================================================
// Stage A: binarize adj -> bf16 A-tiles (WMMA operand layout) + degrees
// One block per 16-row strip; fully coalesced 256-thread column sweeps.
// =====================================================================
__global__ __launch_bounds__(256) void k_binarize_deg(
    const float* __restrict__ adj, __bf16* __restrict__ Atiles,
    float* __restrict__ dvec)
{
    __shared__ float red[256];
    const int t  = threadIdx.x;
    const int mt = blockIdx.x;
    const __bf16 one = (__bf16)1.0f, zero = (__bf16)0.0f;
    float cnt[16];
#pragma unroll
    for (int m = 0; m < 16; ++m) cnt[m] = 0.f;

    for (int m = 0; m < 16; ++m) {
        const int row = mt * 16 + m;
        const float* ap = adj + (size_t)row * NNODE;
        for (int c = t; c < NNODE; c += 256) {
            float v = ap[c];
            bool on = (v > 0.5f) || (c == row);   // binarize + forced diagonal
            cnt[m] += on ? 1.f : 0.f;
            int kt = c >> 5, kin = c & 31;
            Atiles[((size_t)mt * KTILES + kt) * 512 + tile_off(m, kin)] = on ? one : zero;
        }
    }
    for (int m = 0; m < 16; ++m) {
        red[t] = cnt[m];
        __syncthreads();
        for (int s = 128; s > 0; s >>= 1) {
            if (t < s) red[t] += red[t + s];
            __syncthreads();
        }
        if (t == 0) dvec[mt * 16 + m] = rsqrtf(red[0]);
        __syncthreads();
    }
}

// =====================================================================
// Stage B: H1 = X @ W1^T + b1 ; write Hs1 = d*H1 as bf16 B-tiles (NT=8)
// 256 blocks x 32 nodes, LDS-tiled f32 GEMM (k-chunks of 64).
// =====================================================================
__global__ __launch_bounds__(256) void k_dense1(
    const float* __restrict__ X, const float* __restrict__ W1,
    const float* __restrict__ b1, const float* __restrict__ dvec,
    __bf16* __restrict__ Bt)
{
    __shared__ float sW[128][65];
    __shared__ float sX[32][65];
    const int t  = threadIdx.x;
    const int nb = blockIdx.x;
    float acc[16];
#pragma unroll
    for (int i = 0; i < 16; ++i) acc[i] = 0.f;

    for (int kc = 0; kc < INCH / 64; ++kc) {
        for (int l = t; l < 128 * 64; l += 256) {
            int ch = l >> 6, k = l & 63;
            sW[ch][k] = W1[(size_t)ch * INCH + kc * 64 + k];
        }
        for (int l = t; l < 32 * 64; l += 256) {
            int nd = l >> 6, k = l & 63;
            sX[nd][k] = X[(size_t)(nb * 32 + nd) * INCH + kc * 64 + k];
        }
        __syncthreads();
#pragma unroll
        for (int i = 0; i < 16; ++i) {
            int p = t + 256 * i;
            int nd = p >> 7, ch = p & 127;
            float a = 0.f;
#pragma unroll 8
            for (int k = 0; k < 64; ++k) a += sX[nd][k] * sW[ch][k];
            acc[i] += a;
        }
        __syncthreads();
    }
#pragma unroll
    for (int i = 0; i < 16; ++i) {
        int p = t + 256 * i;
        int nd = p >> 7, ch = p & 127;
        int node = nb * 32 + nd;
        float hs = dvec[node] * (acc[i] + b1[ch]);   // fold d_j into B operand
        int kt = node >> 5, kin = node & 31;
        int nt = ch >> 4,   nin = ch & 15;
        Bt[((size_t)kt * 8 + nt) * 512 + tile_off(nin, kin)] = (__bf16)hs;
    }
}

// =====================================================================
// SpMM: partial[ks] = A(bf16) @ Hs(bf16), K-split, one wave32 per block.
// NT = total 16-col B tiles (8 or 4); each wave does 4 of them so one
// A-tile load feeds 4 WMMAs. KTS = K-tiles per split slice.
// =====================================================================
template <int NT, int KTS>
__global__ __launch_bounds__(32) void k_spmm(
    const __bf16* __restrict__ Atiles, const __bf16* __restrict__ Btiles,
    float* __restrict__ partial)
{
    const int lane = threadIdx.x;
    const int mt = blockIdx.x;
    const int nh = blockIdx.y;   // which group of 4 N-tiles
    const int ks = blockIdx.z;   // K slice

    v8f a0 = {}, a1 = {}, a2 = {}, a3 = {};
    const __bf16* ap  = Atiles + (size_t)mt * KTILES * 512 + lane * 16;
    const __bf16* bp0 = Btiles + (size_t)(nh * 4) * 512 + lane * 16;

    const int kt0 = ks * KTS;
    for (int kt = kt0; kt < kt0 + KTS; ++kt) {
        __builtin_prefetch(ap + (size_t)(kt + 2) * 512, 0, 1);   // global_prefetch
        v16bf av = *(const v16bf*)(ap + (size_t)kt * 512);
        const __bf16* bp = bp0 + (size_t)kt * (NT * 512);
        v16bf b0 = *(const v16bf*)(bp);
        v16bf b1 = *(const v16bf*)(bp + 512);
        v16bf b2 = *(const v16bf*)(bp + 1024);
        v16bf b3 = *(const v16bf*)(bp + 1536);
        a0 = __builtin_amdgcn_wmma_f32_16x16x32_bf16(false, av, false, b0, (short)0, a0, false, false);
        a1 = __builtin_amdgcn_wmma_f32_16x16x32_bf16(false, av, false, b1, (short)0, a1, false, false);
        a2 = __builtin_amdgcn_wmma_f32_16x16x32_bf16(false, av, false, b2, (short)0, a2, false, false);
        a3 = __builtin_amdgcn_wmma_f32_16x16x32_bf16(false, av, false, b3, (short)0, a3, false, false);
    }
    // D layout: element (m,n) -> vgpr m&7, lane n + 16*(m>=8)
    const int m0 = (lane >> 4) * 8;
    const int n  = lane & 15;
    const int NCHL = NT * 16;
    float* pout = partial + (size_t)ks * NNODE * NCHL;
#pragma unroll
    for (int v = 0; v < 8; ++v) {
        size_t rb = (size_t)(mt * 16 + m0 + v) * NCHL + n;
        pout[rb + (size_t)(nh * 4 + 0) * 16] = a0[v];
        pout[rb + (size_t)(nh * 4 + 1) * 16] = a1[v];
        pout[rb + (size_t)(nh * 4 + 2) * 16] = a2[v];
        pout[rb + (size_t)(nh * 4 + 3) * 16] = a3[v];
    }
}

// =====================================================================
// Finish layer 1: H2 = relu(d_i * sum_ks partial1)   (f32, [N,128])
// =====================================================================
__global__ __launch_bounds__(256) void k_finish1(
    const float* __restrict__ partial, const float* __restrict__ dvec,
    float* __restrict__ H2)
{
    int tid = blockIdx.x * 256 + threadIdx.x;
    if (tid >= NNODE * HIDCH) return;
    int i = tid >> 7;
    const size_t S = (size_t)NNODE * HIDCH;
    float s = partial[tid] + partial[tid + S] + partial[tid + 2 * S] + partial[tid + 3 * S];
    H2[tid] = fmaxf(0.f, dvec[i] * s);
}

// =====================================================================
// Stage D: H3 = H2 @ W2^T + b2 ; write Hs3 = d*H3 as bf16 B-tiles (NT=4)
// =====================================================================
__global__ __launch_bounds__(256) void k_dense2(
    const float* __restrict__ H2, const float* __restrict__ W2,
    const float* __restrict__ b2, const float* __restrict__ dvec,
    __bf16* __restrict__ Bt)
{
    __shared__ float sW[64][65];
    __shared__ float sH[64][65];
    const int t  = threadIdx.x;
    const int nb = blockIdx.x;   // 128 blocks x 64 nodes
    float acc[16];
#pragma unroll
    for (int i = 0; i < 16; ++i) acc[i] = 0.f;

    for (int kc = 0; kc < HIDCH / 64; ++kc) {
        for (int l = t; l < 64 * 64; l += 256) {
            int ch = l >> 6, k = l & 63;
            sW[ch][k] = W2[(size_t)ch * HIDCH + kc * 64 + k];
        }
        for (int l = t; l < 64 * 64; l += 256) {
            int nd = l >> 6, k = l & 63;
            sH[nd][k] = H2[(size_t)(nb * 64 + nd) * HIDCH + kc * 64 + k];
        }
        __syncthreads();
#pragma unroll
        for (int i = 0; i < 16; ++i) {
            int p = t + 256 * i;
            int nd = p >> 6, ch = p & 63;
            float a = 0.f;
#pragma unroll 8
            for (int k = 0; k < 64; ++k) a += sH[nd][k] * sW[ch][k];
            acc[i] += a;
        }
        __syncthreads();
    }
#pragma unroll
    for (int i = 0; i < 16; ++i) {
        int p = t + 256 * i;
        int nd = p >> 6, ch = p & 63;
        int node = nb * 64 + nd;
        float hs = dvec[node] * (acc[i] + b2[ch]);
        int kt = node >> 5, kin = node & 31;
        int nt = ch >> 4,   nin = ch & 15;
        Bt[((size_t)kt * 4 + nt) * 512 + tile_off(nin, kin)] = (__bf16)hs;
    }
}

// =====================================================================
// Finish layer 2 + gather: out[q] = relu(d_i * sum_ks partial2[i]), i=idx[q]
// =====================================================================
__global__ __launch_bounds__(256) void k_finish2(
    const float* __restrict__ partial, const float* __restrict__ dvec,
    const int* __restrict__ idx, float* __restrict__ out, int nq)
{
    int tid = blockIdx.x * 256 + threadIdx.x;
    if (tid >= nq * OUTCH) return;
    int q = tid >> 6, c = tid & 63;
    int i = idx[q];
    size_t base = (size_t)i * OUTCH + c;
    float s = 0.f;
#pragma unroll
    for (int ks = 0; ks < 8; ++ks)
        s += partial[(size_t)ks * ((size_t)NNODE * OUTCH) + base];
    out[tid] = fmaxf(0.f, dvec[i] * s);
}

// =====================================================================
extern "C" void kernel_launch(void* const* d_in, const int* in_sizes, int n_in,
                              void* d_out, int out_size, void* d_ws, size_t ws_size,
                              hipStream_t stream) {
    const float* adj = (const float*)d_in[0];
    const float* X   = (const float*)d_in[1];
    const float* W1  = (const float*)d_in[2];
    const float* b1  = (const float*)d_in[3];
    const float* W2  = (const float*)d_in[4];
    const float* b2  = (const float*)d_in[5];
    const int*   idx = (const int*)d_in[6];   // jax default x64-off -> int32
    const int nq = in_sizes[6];

    // workspace carve (all offsets 1KB-aligned); total ~167 MB
    char* w = (char*)d_ws;
    float*  dvec = (float*)w;  w += 32 * 1024;                       // [8192] f32
    __bf16* At   = (__bf16*)w; w += (size_t)NNODE * NNODE * 2;       // 128 MB bf16 A tiles
    __bf16* Hs1  = (__bf16*)w; w += (size_t)NNODE * HIDCH * 2;       // 2 MB
    float*  P1   = (float*)w;  w += (size_t)4 * NNODE * HIDCH * 4;   // 16 MB
    float*  H2   = (float*)w;  w += (size_t)NNODE * HIDCH * 4;       // 4 MB
    __bf16* Hs3  = (__bf16*)w; w += (size_t)NNODE * OUTCH * 2;       // 1 MB
    float*  P2   = (float*)w;  w += (size_t)8 * NNODE * OUTCH * 4;   // 16 MB

    k_binarize_deg<<<MTILES, 256, 0, stream>>>(adj, At, dvec);
    k_dense1<<<NNODE / 32, 256, 0, stream>>>(X, W1, b1, dvec, Hs1);
    k_spmm<8, 64><<<dim3(MTILES, 2, 4), 32, 0, stream>>>(At, Hs1, P1);
    k_finish1<<<(NNODE * HIDCH + 255) / 256, 256, 0, stream>>>(P1, dvec, H2);
    k_dense2<<<NNODE / 64, 256, 0, stream>>>(H2, W2, b2, dvec, Hs3);
    k_spmm<4, 32><<<dim3(MTILES, 1, 8), 32, 0, stream>>>(At, Hs3, P2);
    k_finish2<<<(nq * OUTCH + 255) / 256, 256, 0, stream>>>(P2, dvec, idx,
                                                            (float*)d_out, nq);
    (void)n_in; (void)out_size; (void)ws_size;
}
